// Qwen3SteeringAttention_64622077936221
// MI455X (gfx1250) — matrix operand order repaired and checked by reading, not compiled, not run on hardware
//
#include <hip/hip_runtime.h>

// ---------------------------------------------------------------------------
// Problem constants
// ---------------------------------------------------------------------------
#define BATCH   2
#define SEQ     2048
#define HID     2048
#define NH      16
#define NKV     8
#define HD      128
#define GROUPS  2
#define MROWS   4096          // BATCH*SEQ
#define SCALING 0.08838834764831845f

typedef __attribute__((ext_vector_type(16))) __bf16 bf16x16;
typedef __attribute__((ext_vector_type(8)))  __bf16 bf16x8;
typedef __attribute__((ext_vector_type(8)))  float  f32x8;

typedef unsigned int v4u __attribute__((ext_vector_type(4)));
typedef int v8i __attribute__((ext_vector_type(8)));
typedef int v4i __attribute__((ext_vector_type(4)));

// addrspace-qualified int4 pointee types for the async-LDS builtins
typedef __attribute__((address_space(1))) v4i as1_v4i;
typedef __attribute__((address_space(3))) v4i as3_v4i;
typedef void __attribute__((address_space(3))) as3_void;

union Frag16 { bf16x16 v; bf16x8 h[2]; };

// ---------------------------------------------------------------------------
// CDNA5 feature detection (compile-safe: fall back to sync paths if absent)
// ---------------------------------------------------------------------------
#if __has_builtin(__builtin_amdgcn_tensor_load_to_lds)
#define HAVE_TDM 1
#endif
#if __has_builtin(__builtin_amdgcn_global_load_async_to_lds_b128)
#define HAVE_ASYNC 1
#endif

__device__ __forceinline__ unsigned lds_offset_of(const void* p) {
    return (unsigned)(uintptr_t)(const as3_void*)p;
}

__device__ __forceinline__ void wait_tensorcnt0() {
#if __has_builtin(__builtin_amdgcn_s_wait_tensorcnt)
    __builtin_amdgcn_s_wait_tensorcnt(0);
#else
    asm volatile("s_wait_tensorcnt 0" ::: "memory");
#endif
}

__device__ __forceinline__ void wait_asynccnt0() {
#if __has_builtin(__builtin_amdgcn_s_wait_asynccnt)
    __builtin_amdgcn_s_wait_asynccnt(0);
#else
    asm volatile("s_wait_asynccnt 0" ::: "memory");
#endif
}

#ifdef HAVE_TDM
// Issue a 2-D TDM load of bf16 data: tile_d1 rows x tile_d0 elems, row stride
// stride0 (elements), from gsrc into LDS at byte offset lds_off.
// Descriptor layout per CDNA5 ISA (D# groups 0/1).
// 6-arg builtin form (amdgpu-toolchain clang-23 / therock-10.0).
__device__ __forceinline__ void tdm_load_2d_bf16(const __bf16* gsrc,
                                                 unsigned lds_off,
                                                 unsigned tensor_d0,
                                                 unsigned tensor_d1,
                                                 unsigned tile_d0,
                                                 unsigned tile_d1,
                                                 unsigned stride0) {
    unsigned long long ga = (unsigned long long)(uintptr_t)gsrc;
    v4u g0;
    g0[0] = 1u;                                    // count=1 (valid user D#)
    g0[1] = lds_off;                               // lds_addr (bytes)
    g0[2] = (unsigned)(ga & 0xFFFFFFFFu);          // global_addr[31:0]
    g0[3] = (unsigned)((ga >> 32) & 0x01FFFFFFu)   // global_addr[56:32]
          | 0x80000000u;                           // type=2 ("image")
    v8i g1;
    g1[0] = (int)(1u << 16);                       // wg_mask=0, data_size=1 (2B)
    g1[1] = (int)((tensor_d0 & 0xFFFFu) << 16);    // tensor_dim0[15:0]
    g1[2] = (int)((tensor_d0 >> 16) | ((tensor_d1 & 0xFFFFu) << 16));
    g1[3] = (int)((tensor_d1 >> 16) | (tile_d0 << 16));   // tile_dim0
    g1[4] = (int)(tile_d1 & 0xFFFFu);              // tile_dim1, tile_dim2=0
    g1[5] = (int)stride0;                          // tensor_dim0_stride[31:0]
    g1[6] = 0;                                     // stride0 hi, stride1 lo
    g1[7] = 0;                                     // stride1 hi
    v4i g2 = {0, 0, 0, 0};
    v4i g3 = {0, 0, 0, 0};
    v8i g4 = {0, 0, 0, 0, 0, 0, 0, 0};
    __builtin_amdgcn_tensor_load_to_lds(g0, g1, g2, g3, g4, 0);
}
#endif

__device__ __forceinline__ f32x8 wmma_bf16(bf16x16 a, bf16x16 b, f32x8 c) {
    // D = A(16x32 bf16) * B(32x16 bf16) + C(16x16 f32)
    return __builtin_amdgcn_wmma_f32_16x16x32_bf16(
        /*neg_a=*/false, a, /*neg_b=*/false, b,
        /*c_mod=*/(short)0, c, /*reuse_a=*/false, /*reuse_b=*/false);
}

__device__ __forceinline__ f32x8 zero_f32x8() {
    f32x8 z = {0.f, 0.f, 0.f, 0.f, 0.f, 0.f, 0.f, 0.f};
    return z;
}

// ---------------------------------------------------------------------------
// fp32 -> bf16 conversion
// ---------------------------------------------------------------------------
__global__ void convert_f32_bf16_kernel(const float* __restrict__ src,
                                        __bf16* __restrict__ dst, int n) {
    int i = blockIdx.x * blockDim.x + threadIdx.x;
    if (i < n) dst[i] = (__bf16)src[i];
}

// ---------------------------------------------------------------------------
// Steering statistics: p, gamma, eta, and fused per-(head,key) logit bias.
// bias[h][k] = clip(log(p),-2,2)*hm1[h] + hm2[h]*0.5*log(p*S)
// ---------------------------------------------------------------------------
__global__ void stats_kernel(const float* __restrict__ prior,
                             const float* __restrict__ hm1,
                             const float* __restrict__ hm2,
                             float* __restrict__ gamma,
                             float* __restrict__ eta,
                             float* __restrict__ keybias) {
    __shared__ float sred[256];
    int t = threadIdx.x;
    float acc = 0.f;
    for (int k = t; k < SEQ; k += 256) acc += prior[k];
    sred[t] = acc;
    __syncthreads();
    for (int off = 128; off > 0; off >>= 1) {
        if (t < off) sred[t] += sred[t + off];
        __syncthreads();
    }
    float total = sred[0];
    for (int k = t; k < SEQ; k += 256) {
        float p = prior[k] / total;
        float c = p * (float)SEQ - 1.0f;
        float g = fminf(fmaxf(1.0f + 0.5f * c, 0.8f), 1.25f);
        float e = fminf(fmaxf(1.0f + 0.5f * c, 0.9f), 1.1f);
        float delta = fminf(fmaxf(logf(p + 1e-8f), -2.0f), 2.0f);
        float lps = 0.5f * logf(p * (float)SEQ);
        gamma[k] = g;
        eta[k] = e;
        for (int h = 0; h < NH; ++h)
            keybias[h * SEQ + k] = delta * hm1[h] + hm2[h] * lps;
    }
}

// ---------------------------------------------------------------------------
// bf16 WMMA GEMM: C[M,N](f32) = A[M,K](bf16, row-major) * B[K,N](bf16, row-major)
// block = 256 threads (8 waves), tile 64(M) x 128(N), K-step 32.
// A tile staged with global_load_async_to_lds_b128 when available.
// ---------------------------------------------------------------------------
__global__ __launch_bounds__(256) void gemm_bf16_kernel(
    const __bf16* __restrict__ A, const __bf16* __restrict__ Bw,
    float* __restrict__ C, int M, int N, int K) {
    __shared__ __bf16 As[64][40];    // [m][k] padded
    __shared__ __bf16 Bs[128][40];   // [n][k] padded (transposed)

    int n0 = blockIdx.x * 128;
    int m0 = blockIdx.y * 64;
    int tid = threadIdx.x;
    int wave = tid >> 5, lane = tid & 31;
    int lm = lane & 15, hh = lane >> 4;
    int wm = (wave >> 2) * 32;
    int wn = (wave & 3) * 32;

    f32x8 acc[2][2];
#pragma unroll
    for (int mi = 0; mi < 2; ++mi)
#pragma unroll
        for (int ni = 0; ni < 2; ++ni) acc[mi][ni] = zero_f32x8();

    int arow = tid >> 2, ach = (tid & 3) * 8;    // A: 64 rows x 4 chunks of 8
    int brow = tid >> 3, bch = (tid & 7) * 16;   // B: 32 rows x 8 chunks of 16

    const __bf16* ga = A + (size_t)(m0 + arow) * K + ach;
    const __bf16* gb = Bw + (size_t)brow * N + n0 + bch;

    for (int kk = 0; kk < K; kk += 32) {
        // stage A tile (async LDS DMA when the toolchain exposes it)
#ifdef HAVE_ASYNC
        __builtin_amdgcn_global_load_async_to_lds_b128(
            (as1_v4i*)(ga + kk), (as3_v4i*)&As[arow][ach], 0, 0);
#else
        {
            bf16x8 av = *reinterpret_cast<const bf16x8*>(ga + kk);
            *reinterpret_cast<bf16x8*>(&As[arow][ach]) = av;
        }
#endif
        // stage B tile (transposed into N-major)
        bf16x8 bv0 = *reinterpret_cast<const bf16x8*>(gb + (size_t)kk * N);
        bf16x8 bv1 = *reinterpret_cast<const bf16x8*>(gb + (size_t)kk * N + 8);
#pragma unroll
        for (int j = 0; j < 8; ++j) {
            Bs[bch + j][brow] = bv0[j];
            Bs[bch + 8 + j][brow] = bv1[j];
        }
        if (kk + 32 < K) {
            __builtin_prefetch(ga + kk + 32, 0, 1);
            __builtin_prefetch(gb + (size_t)(kk + 32) * N, 0, 1);
        }
#ifdef HAVE_ASYNC
        wait_asynccnt0();
#endif
        __syncthreads();

        Frag16 af[2], bfr[2];
#pragma unroll
        for (int mi = 0; mi < 2; ++mi) {
            int row = wm + mi * 16 + lm;
            af[mi].h[0] = *reinterpret_cast<const bf16x8*>(&As[row][hh * 8]);
            af[mi].h[1] = *reinterpret_cast<const bf16x8*>(&As[row][16 + hh * 8]);
        }
#pragma unroll
        for (int ni = 0; ni < 2; ++ni) {
            int col = wn + ni * 16 + lm;
            bfr[ni].h[0] = *reinterpret_cast<const bf16x8*>(&Bs[col][hh * 8]);
            bfr[ni].h[1] = *reinterpret_cast<const bf16x8*>(&Bs[col][16 + hh * 8]);
        }
#pragma unroll
        for (int mi = 0; mi < 2; ++mi)
#pragma unroll
            for (int ni = 0; ni < 2; ++ni)
                acc[mi][ni] = wmma_bf16(af[mi].v, bfr[ni].v, acc[mi][ni]);
        __syncthreads();
    }

#pragma unroll
    for (int mi = 0; mi < 2; ++mi)
#pragma unroll
        for (int ni = 0; ni < 2; ++ni)
#pragma unroll
            for (int r = 0; r < 8; ++r) {
                int row = m0 + wm + mi * 16 + r + 8 * hh;
                int col = n0 + wn + ni * 16 + lm;
                C[(size_t)row * N + col] = acc[mi][ni][r];
            }
}

// ---------------------------------------------------------------------------
// Q/K epilogue: per-head RMSNorm + RoPE (+ optional gamma key scale),
// fp32 (b,s,h,d) -> bf16 (b,h,s,d)
// ---------------------------------------------------------------------------
__global__ __launch_bounds__(128) void epi_qk_kernel(
    const float* __restrict__ Xf, const float* __restrict__ nw,
    const float* __restrict__ cosp, const float* __restrict__ sinp,
    const float* __restrict__ scale_s, __bf16* __restrict__ out, int nheads) {
    __shared__ float red[128];
    __shared__ float xs[128];
    int idx = blockIdx.x;
    int hh = idx % nheads;
    int s = (idx / nheads) % SEQ;
    int b = idx / (nheads * SEQ);
    int d = threadIdx.x;

    float x = Xf[(((size_t)(b * SEQ + s)) * nheads + hh) * HD + d];
    red[d] = x * x;
    __syncthreads();
    for (int off = 64; off > 0; off >>= 1) {
        if (d < off) red[d] += red[d + off];
        __syncthreads();
    }
    float rs = rsqrtf(red[0] * (1.0f / (float)HD) + 1e-6f);
    float xn = x * rs * nw[d];
    xs[d] = xn;
    __syncthreads();
    float rot = (d < 64) ? -xs[d + 64] : xs[d - 64];
    size_t ci = ((size_t)(b * SEQ + s)) * HD + d;
    float val = xn * cosp[ci] + rot * sinp[ci];
    if (scale_s) val *= scale_s[s];
    out[(((size_t)(b * nheads + hh)) * SEQ + s) * HD + d] = (__bf16)val;
}

// ---------------------------------------------------------------------------
// V epilogue: eta scale, fp32 (b,s,kv,d) -> bf16 (b,kv,s,d)
// ---------------------------------------------------------------------------
__global__ void epi_v_kernel(const float* __restrict__ Vf,
                             const float* __restrict__ eta,
                             __bf16* __restrict__ out) {
    int i = blockIdx.x * blockDim.x + threadIdx.x;
    const int total = BATCH * SEQ * NKV * HD;
    if (i >= total) return;
    int d = i & (HD - 1);
    int kv = (i >> 7) & (NKV - 1);
    int s = (i / (HD * NKV)) % SEQ;
    int b = i / (HD * NKV * SEQ);
    float v = Vf[(((size_t)(b * SEQ + s)) * NKV + kv) * HD + d] * eta[s];
    out[(((size_t)(b * NKV + kv)) * SEQ + s) * HD + d] = (__bf16)v;
}

// ---------------------------------------------------------------------------
// Flash attention, WMMA bf16. Block = 4 waves, 64 q-rows, key blocks of 32.
// K tile staged into LDS via the Tensor Data Mover (TENSORcnt) when available.
// ---------------------------------------------------------------------------
__global__ __launch_bounds__(128) void flash_attn_kernel(
    const __bf16* __restrict__ Qb, const __bf16* __restrict__ Kb,
    const __bf16* __restrict__ Vb, const float* __restrict__ keybias,
    __bf16* __restrict__ Ob) {
    __shared__ __bf16 Kl[32][128];     // K tile, row-major [key][d]   (8 KB)
    __shared__ __bf16 Pl[4][16][40];   // per-wave P tile (16q x 32k), padded
    __shared__ __bf16 Vl[128][40];     // V transposed [d][key], padded

    int b = blockIdx.z, h = blockIdx.y;
    int q0 = blockIdx.x * 64;
    int kvh = h / GROUPS;
    int tid = threadIdx.x;
    int wave = tid >> 5, lane = tid & 31;
    int lm = lane & 15, hh = lane >> 4;
    int qbase = q0 + wave * 16;

    const __bf16* Qp = Qb + ((size_t)(b * NH + h) * SEQ) * HD;
    const __bf16* Kp = Kb + ((size_t)(b * NKV + kvh) * SEQ) * HD;
    const __bf16* Vp = Vb + ((size_t)(b * NKV + kvh) * SEQ) * HD;
    const float* kbh = keybias + (size_t)h * SEQ;

    // Q A-fragments held in registers for the whole key loop (4 x 32-k chunks)
    Frag16 qfrag[4];
#pragma unroll
    for (int ks = 0; ks < 4; ++ks) {
        const __bf16* qp = Qp + (size_t)(qbase + lm) * HD + ks * 32;
        qfrag[ks].h[0] = *reinterpret_cast<const bf16x8*>(qp + hh * 8);
        qfrag[ks].h[1] = *reinterpret_cast<const bf16x8*>(qp + 16 + hh * 8);
    }

    f32x8 oacc[8];
#pragma unroll
    for (int t = 0; t < 8; ++t) oacc[t] = zero_f32x8();
    float mrow[8], lrow[8];
#pragma unroll
    for (int r = 0; r < 8; ++r) { mrow[r] = -1e30f; lrow[r] = 0.f; }

#ifdef HAVE_TDM
    unsigned kl_off = lds_offset_of(&Kl[0][0]);
#endif

    int kend = q0 + 64;   // causal: keys < q0+64
    for (int k0 = 0; k0 < kend; k0 += 32) {
        // ---- stage K tile (32 keys x 128 d) into LDS ----
#ifdef HAVE_TDM
        if (wave == 0) {
            tdm_load_2d_bf16(Kp + (size_t)k0 * HD, kl_off,
                             /*tensor_d0=*/HD, /*tensor_d1=*/SEQ,
                             /*tile_d0=*/HD, /*tile_d1=*/32, /*stride0=*/HD);
            wait_tensorcnt0();
        }
#else
        {
            int key = tid >> 2, dch = tid & 3;
            const __bf16* ksrc = Kp + (size_t)(k0 + key) * HD + dch * 32;
#ifdef HAVE_ASYNC
#pragma unroll
            for (int c = 0; c < 4; ++c)
                __builtin_amdgcn_global_load_async_to_lds_b128(
                    (as1_v4i*)(ksrc + c * 8),
                    (as3_v4i*)&Kl[key][dch * 32 + c * 8], 0, 0);
            wait_asynccnt0();
#else
#pragma unroll
            for (int c = 0; c < 4; ++c) {
                bf16x8 kv8 = *reinterpret_cast<const bf16x8*>(ksrc + c * 8);
                *reinterpret_cast<bf16x8*>(&Kl[key][dch * 32 + c * 8]) = kv8;
            }
#endif
        }
#endif
        // ---- cooperative V load + transpose [key][d] -> [d][key] ----
        {
            int key = tid >> 2, dch = tid & 3;
            const __bf16* vp = Vp + (size_t)(k0 + key) * HD + dch * 32;
#pragma unroll
            for (int c = 0; c < 4; ++c) {
                bf16x8 v8 = *reinterpret_cast<const bf16x8*>(vp + c * 8);
#pragma unroll
                for (int j = 0; j < 8; ++j) Vl[dch * 32 + c * 8 + j][key] = v8[j];
            }
        }
        __syncthreads();    // Kl + Vl published

        // ---- S = Q * K^T (two 16-key n-tiles, 4 d-chunks each) ----
        f32x8 sacc0 = zero_f32x8(), sacc1 = zero_f32x8();
#pragma unroll
        for (int ks = 0; ks < 4; ++ks) {
            {
                const __bf16* kp = &Kl[lm][ks * 32 + hh * 8];
                Frag16 f;
                f.h[0] = *reinterpret_cast<const bf16x8*>(kp);
                f.h[1] = *reinterpret_cast<const bf16x8*>(kp + 16);
                sacc0 = wmma_bf16(qfrag[ks].v, f.v, sacc0);
            }
            {
                const __bf16* kp = &Kl[16 + lm][ks * 32 + hh * 8];
                Frag16 f;
                f.h[0] = *reinterpret_cast<const bf16x8*>(kp);
                f.h[1] = *reinterpret_cast<const bf16x8*>(kp + 16);
                sacc1 = wmma_bf16(qfrag[ks].v, f.v, sacc1);
            }
        }

        // ---- bias + causal mask + online softmax (half-wave reductions) ----
        float p0[8], p1[8], rowscale[8];
#pragma unroll
        for (int r = 0; r < 8; ++r) {
            int row = qbase + r + 8 * hh;
            int c0 = k0 + lm, c1 = k0 + 16 + lm;
            float v0 = sacc0[r] * SCALING + kbh[c0];
            float v1 = sacc1[r] * SCALING + kbh[c1];
            if (c0 > row) v0 = -1e30f;
            if (c1 > row) v1 = -1e30f;
            float mx = fmaxf(v0, v1);
#pragma unroll
            for (int o = 1; o < 16; o <<= 1) mx = fmaxf(mx, __shfl_xor(mx, o, 16));
            float mnew = fmaxf(mrow[r], mx);
            float e0 = __expf(v0 - mnew);
            float e1 = __expf(v1 - mnew);
            float rsum = e0 + e1;
#pragma unroll
            for (int o = 1; o < 16; o <<= 1) rsum += __shfl_xor(rsum, o, 16);
            float alpha = __expf(mrow[r] - mnew);
            lrow[r] = lrow[r] * alpha + rsum;
            mrow[r] = mnew;
            rowscale[r] = alpha;
            p0[r] = e0;
            p1[r] = e1;
        }
#pragma unroll
        for (int t = 0; t < 8; ++t)
#pragma unroll
            for (int r = 0; r < 8; ++r) oacc[t][r] *= rowscale[r];

        // ---- P (C-layout) -> LDS (row-major) for A-fragment reload ----
#pragma unroll
        for (int r = 0; r < 8; ++r) {
            Pl[wave][r + 8 * hh][lm] = (__bf16)p0[r];
            Pl[wave][r + 8 * hh][lm + 16] = (__bf16)p1[r];
        }
        __syncthreads();    // Pl published

        // ---- O += P * V ----
        Frag16 pf;
        pf.h[0] = *reinterpret_cast<const bf16x8*>(&Pl[wave][lm][hh * 8]);
        pf.h[1] = *reinterpret_cast<const bf16x8*>(&Pl[wave][lm][16 + hh * 8]);
#pragma unroll
        for (int t = 0; t < 8; ++t) {
            Frag16 vf;
            vf.h[0] = *reinterpret_cast<const bf16x8*>(&Vl[t * 16 + lm][hh * 8]);
            vf.h[1] = *reinterpret_cast<const bf16x8*>(&Vl[t * 16 + lm][16 + hh * 8]);
            oacc[t] = wmma_bf16(pf.v, vf.v, oacc[t]);
        }
        __syncthreads();    // protect Kl/Pl/Vl before next iteration's stores
    }

    // ---- normalize and write (b, s, h, d) bf16 ----
#pragma unroll
    for (int t = 0; t < 8; ++t)
#pragma unroll
        for (int r = 0; r < 8; ++r) {
            int row = qbase + r + 8 * hh;
            int col = t * 16 + lm;
            float val = oacc[t][r] / lrow[r];
            Ob[(((size_t)(b * SEQ + row)) * NH + h) * HD + col] = (__bf16)val;
        }
}

// ---------------------------------------------------------------------------
// Host launcher
// ---------------------------------------------------------------------------
extern "C" void kernel_launch(void* const* d_in, const int* in_sizes, int n_in,
                              void* d_out, int out_size, void* d_ws, size_t ws_size,
                              hipStream_t stream) {
    (void)in_sizes; (void)n_in; (void)out_size; (void)ws_size;
    const float* hs    = (const float*)d_in[0];
    const float* cosp  = (const float*)d_in[1];
    const float* sinp  = (const float*)d_in[2];
    // d_in[3] = attention_mask (causal; applied analytically)
    const float* prior = (const float*)d_in[4];
    const float* hm1   = (const float*)d_in[5];
    const float* hm2   = (const float*)d_in[6];
    const float* Wq    = (const float*)d_in[7];
    const float* Wk    = (const float*)d_in[8];
    const float* Wv    = (const float*)d_in[9];
    const float* Wo    = (const float*)d_in[10];
    const float* qnw   = (const float*)d_in[11];
    const float* knw   = (const float*)d_in[12];
    float* out = (float*)d_out;

    char* ws = (char*)d_ws;
    size_t off = 0;
    auto take = [&](size_t bytes) -> char* {
        char* p = ws + off;
        off += (bytes + 255) & ~(size_t)255;
        return p;
    };
    __bf16* hb  = (__bf16*)take((size_t)MROWS * HID * 2);
    __bf16* wqb = (__bf16*)take((size_t)HID * 2048 * 2);
    __bf16* wkb = (__bf16*)take((size_t)HID * 1024 * 2);
    __bf16* wvb = (__bf16*)take((size_t)HID * 1024 * 2);
    __bf16* wob = (__bf16*)take((size_t)2048 * HID * 2);
    float*  qf  = (float*)take((size_t)MROWS * 2048 * 4);
    float*  kf  = (float*)take((size_t)MROWS * 1024 * 4);
    float*  vf  = (float*)take((size_t)MROWS * 1024 * 4);
    __bf16* qb  = (__bf16*)take((size_t)MROWS * 2048 * 2);
    __bf16* kb  = (__bf16*)take((size_t)MROWS * 1024 * 2);
    __bf16* vb  = (__bf16*)take((size_t)MROWS * 1024 * 2);
    __bf16* ob  = (__bf16*)take((size_t)MROWS * 2048 * 2);
    float*  gam = (float*)take((size_t)SEQ * 4);
    float*  eta = (float*)take((size_t)SEQ * 4);
    float*  kbias = (float*)take((size_t)NH * SEQ * 4);

    auto cvt = [&](const float* s, __bf16* d, int n) {
        convert_f32_bf16_kernel<<<(n + 255) / 256, 256, 0, stream>>>(s, d, n);
    };
    cvt(hs, hb, MROWS * HID);
    cvt(Wq, wqb, HID * 2048);
    cvt(Wk, wkb, HID * 1024);
    cvt(Wv, wvb, HID * 1024);
    cvt(Wo, wob, 2048 * HID);

    stats_kernel<<<1, 256, 0, stream>>>(prior, hm1, hm2, gam, eta, kbias);

    // projections (WMMA GEMMs)
    gemm_bf16_kernel<<<dim3(2048 / 128, MROWS / 64), 256, 0, stream>>>(
        hb, wqb, qf, MROWS, 2048, HID);
    gemm_bf16_kernel<<<dim3(1024 / 128, MROWS / 64), 256, 0, stream>>>(
        hb, wkb, kf, MROWS, 1024, HID);
    gemm_bf16_kernel<<<dim3(1024 / 128, MROWS / 64), 256, 0, stream>>>(
        hb, wvb, vf, MROWS, 1024, HID);

    // epilogues: RMSNorm + RoPE (+ gamma for K), eta for V
    epi_qk_kernel<<<BATCH * SEQ * NH, 128, 0, stream>>>(qf, qnw, cosp, sinp,
                                                        nullptr, qb, NH);
    epi_qk_kernel<<<BATCH * SEQ * NKV, 128, 0, stream>>>(kf, knw, cosp, sinp,
                                                         gam, kb, NKV);
    {
        int n = BATCH * SEQ * NKV * HD;
        epi_v_kernel<<<(n + 255) / 256, 256, 0, stream>>>(vf, eta, vb);
    }

    // flash attention (WMMA + TDM)
    flash_attn_kernel<<<dim3(SEQ / 64, NH, BATCH), 128, 0, stream>>>(
        qb, kb, vb, kbias, ob);

    // output projection (WMMA GEMM) -> fp32 result
    gemm_bf16_kernel<<<dim3(HID / 128, MROWS / 64), 256, 0, stream>>>(
        ob, wob, out, MROWS, HID, 2048);
}
